// QCCPN_69372311765581
// MI455X (gfx1250) — matrix operand order repaired and checked by reading, not compiled, not run on hardware
//
#include <hip/hip_runtime.h>
#include <math.h>

#define GAMMA 0.1f
#define B_SZ   4096
#define IN_FT  1024
#define FEAT   512
#define NCLASS 1000
#define NPAD   1008          // 63 * 16, padded class count
#define D_F    (4 * FEAT)    // 2048 concatenated feature width

typedef __bf16 bf16_t;
typedef __attribute__((ext_vector_type(16))) bf16_t v16bf;
typedef __attribute__((ext_vector_type(8)))  bf16_t v8bf;
typedef __attribute__((ext_vector_type(8)))  float  v8f;

union V16BF { v16bf v; bf16_t e[16]; v8bf h[2]; };
union V8BF  { v8bf  v; bf16_t e[8]; };
union V8F   { v8f   v; float  e[8]; };

// ---------------------------------------------------------------------------
// Kernel: bulk fp32 -> bf16 conversion, 8 elements / thread (b128 in, b128 out)
// ---------------------------------------------------------------------------
__global__ __launch_bounds__(256) void cvt_bf16(const float* __restrict__ src,
                                                bf16_t* __restrict__ dst, int n) {
    int i = (blockIdx.x * 256 + threadIdx.x) * 8;
    if (i < n) {
        float4 a = *(const float4*)(src + i);
        float4 b = *(const float4*)(src + i + 4);
        V8BF o;
        o.e[0] = (bf16_t)a.x; o.e[1] = (bf16_t)a.y; o.e[2] = (bf16_t)a.z; o.e[3] = (bf16_t)a.w;
        o.e[4] = (bf16_t)b.x; o.e[5] = (bf16_t)b.y; o.e[6] = (bf16_t)b.z; o.e[7] = (bf16_t)b.w;
        *(v8bf*)(dst + i) = o.v;
    }
}

// ---------------------------------------------------------------------------
// Kernel: bf16-convert centers (zero-pad rows [1000,1008)) and compute ||c||^2
// ---------------------------------------------------------------------------
__global__ __launch_bounds__(256) void prep_centers(const float* __restrict__ centers,
                                                    bf16_t* __restrict__ cbf,
                                                    float* __restrict__ c2) {
    const int row = blockIdx.x;          // 0..NPAD-1
    __shared__ float red[256];
    float ss = 0.0f;
    if (row < NCLASS) {
        for (int k = threadIdx.x; k < D_F; k += 256) {
            float v = centers[(size_t)row * D_F + k];
            cbf[(size_t)row * D_F + k] = (bf16_t)v;
            ss += v * v;
        }
    } else {
        for (int k = threadIdx.x; k < D_F; k += 256)
            cbf[(size_t)row * D_F + k] = (bf16_t)0.0f;
    }
    red[threadIdx.x] = ss;
    __syncthreads();
    for (int s = 128; s > 0; s >>= 1) {
        if (threadIdx.x < s) red[threadIdx.x] += red[threadIdx.x + s];
        __syncthreads();
    }
    if (threadIdx.x == 0) c2[row] = red[0];
}

__global__ __launch_bounds__(256) void zero_f2(float* __restrict__ f2) {
    int i = blockIdx.x * 256 + threadIdx.x;
    if (i < B_SZ) f2[i] = 0.0f;
}

// ---------------------------------------------------------------------------
// Feature GEMM (all-bf16 inputs): feat[:, p*FEAT + n] = relu(x_p @ W_p^T + b_p)
// 4-wave block computes a 128(M) x 64(N) tile of part p = blockIdx.z.
// Each wave: 32x64 (2 A-frags x 4 B-frags = 8 WMMA / K-step).
// Also accumulates row sums-of-squares into f2.
// ---------------------------------------------------------------------------
__global__ __launch_bounds__(128) void feature_gemm(
    const bf16_t* __restrict__ xbf,   // [4][B_SZ][IN_FT] concatenated
    const bf16_t* __restrict__ wbf,   // [2][FEAT][IN_FT]  (W_f then W_r)
    const float* __restrict__ bfv, const float* __restrict__ brv,
    bf16_t* __restrict__ feat, float* __restrict__ f2)
{
    const int wave = threadIdx.x >> 5;   // 0..3
    const int lane = threadIdx.x & 31;
    const int half = lane >> 4;          // K-half selector
    const int ln16 = lane & 15;

    const int p = blockIdx.z;
    const bf16_t* xp = xbf + (size_t)p * B_SZ * IN_FT;
    const bf16_t* Wp = wbf + (size_t)(p != 0) * FEAT * IN_FT;
    const float*  bp = (p == 0) ? bfv : brv;

    const int m0 = blockIdx.x * 128 + wave * 32;  // this wave's 32 rows
    const int n0 = blockIdx.y * 64;               // 64-col tile

    V8F acc[2][4];
    #pragma unroll
    for (int t = 0; t < 2; ++t)
        #pragma unroll
        for (int j = 0; j < 4; ++j)
            #pragma unroll
            for (int r = 0; r < 8; ++r) acc[t][j].e[r] = 0.0f;

    // A fragment: row = m0 + t*16 + ln16; 16 contiguous k at k0 + half*16
    const bf16_t* abase0 = xp + (size_t)(m0 + ln16) * IN_FT + half * 16;
    const bf16_t* abase1 = abase0 + (size_t)16 * IN_FT;
    // B fragment: B[k][n] = W[n][k]; 16 contiguous k at k0 + half*16
    const bf16_t* bbase = Wp + (size_t)(n0 + ln16) * IN_FT + half * 16;

    for (int k0 = 0; k0 < IN_FT; k0 += 32) {
        V16BF a0, a1;
        a0.h[0] = *(const v8bf*)(abase0 + k0);
        a0.h[1] = *(const v8bf*)(abase0 + k0 + 8);
        a1.h[0] = *(const v8bf*)(abase1 + k0);
        a1.h[1] = *(const v8bf*)(abase1 + k0 + 8);
        #pragma unroll
        for (int j = 0; j < 4; ++j) {
            V16BF b;
            b.v = *(const v16bf*)(bbase + (size_t)j * 16 * IN_FT + k0);
            acc[0][j].v = __builtin_amdgcn_wmma_f32_16x16x32_bf16(
                false, a0.v, false, b.v, (short)0, acc[0][j].v, false, false);
            acc[1][j].v = __builtin_amdgcn_wmma_f32_16x16x32_bf16(
                false, a1.v, false, b.v, (short)0, acc[1][j].v, false, false);
        }
    }

    // Epilogue: bias + relu, store bf16 feat, accumulate ||row||^2
    #pragma unroll
    for (int t = 0; t < 2; ++t) {
        float rss[8];
        #pragma unroll
        for (int r = 0; r < 8; ++r) rss[r] = 0.0f;

        #pragma unroll
        for (int j = 0; j < 4; ++j) {
            const int n = n0 + j * 16 + ln16;
            const float bias = bp[n];
            #pragma unroll
            for (int r = 0; r < 8; ++r) {
                float v = acc[t][j].e[r] + bias;
                v = fmaxf(v, 0.0f);
                const int M = m0 + t * 16 + r + 8 * half;
                feat[(size_t)M * D_F + p * FEAT + n] = (bf16_t)v;
                rss[r] += v * v;
            }
        }
        // Lanes ln16=0..15 of a half hold the same M rows: butterfly-reduce
        #pragma unroll
        for (int r = 0; r < 8; ++r) {
            float s = rss[r];
            #pragma unroll
            for (int mask = 1; mask < 16; mask <<= 1)
                s += __shfl_xor(s, mask, 32);
            if (ln16 == 0)
                atomicAdd(&f2[m0 + t * 16 + r + 8 * half], s);
        }
    }
}

// ---------------------------------------------------------------------------
// Distance GEMM: dot = feat @ centers^T (bf16 WMMA, K = 2048),
// out = -gamma * sqrt(max(||f||^2 + ||c||^2 - 2*dot, 0))
// 4-wave block computes 128(M) x 16(N); each wave 32x16 (2 WMMA / K-step).
// ---------------------------------------------------------------------------
__global__ __launch_bounds__(128) void dist_gemm(
    const bf16_t* __restrict__ feat, const bf16_t* __restrict__ cbf,
    const float* __restrict__ f2, const float* __restrict__ c2,
    float* __restrict__ out)
{
    const int wave = threadIdx.x >> 5;
    const int lane = threadIdx.x & 31;
    const int half = lane >> 4;
    const int ln16 = lane & 15;

    const int m0 = blockIdx.x * 128 + wave * 32;
    const int n0 = blockIdx.y * 16;

    V8F acc0, acc1;
    #pragma unroll
    for (int r = 0; r < 8; ++r) { acc0.e[r] = 0.0f; acc1.e[r] = 0.0f; }

    const bf16_t* abase0 = feat + (size_t)(m0 + ln16) * D_F + half * 16;
    const bf16_t* abase1 = abase0 + (size_t)16 * D_F;
    const bf16_t* bbase  = cbf  + (size_t)(n0 + ln16) * D_F + half * 16;

    for (int k0 = 0; k0 < D_F; k0 += 32) {
        V16BF b;
        b.v = *(const v16bf*)(bbase + k0);
        V16BF a0, a1;
        a0.h[0] = *(const v8bf*)(abase0 + k0);
        a0.h[1] = *(const v8bf*)(abase0 + k0 + 8);
        a1.h[0] = *(const v8bf*)(abase1 + k0);
        a1.h[1] = *(const v8bf*)(abase1 + k0 + 8);
        acc0.v = __builtin_amdgcn_wmma_f32_16x16x32_bf16(
            false, a0.v, false, b.v, (short)0, acc0.v, false, false);
        acc1.v = __builtin_amdgcn_wmma_f32_16x16x32_bf16(
            false, a1.v, false, b.v, (short)0, acc1.v, false, false);
    }

    const int n = n0 + ln16;
    const float c2n = (n < NCLASS) ? c2[n] : 0.0f;
    #pragma unroll
    for (int t = 0; t < 2; ++t) {
        const V8F& ac = t ? acc1 : acc0;
        #pragma unroll
        for (int r = 0; r < 8; ++r) {
            const int M = m0 + t * 16 + r + 8 * half;
            float d2 = f2[M] + c2n - 2.0f * ac.e[r];
            float d = sqrtf(fmaxf(d2, 0.0f));
            if (n < NCLASS)
                out[(size_t)M * NCLASS + n] = -GAMMA * d;
        }
    }
}

// ---------------------------------------------------------------------------
extern "C" void kernel_launch(void* const* d_in, const int* in_sizes, int n_in,
                              void* d_out, int out_size, void* d_ws, size_t ws_size,
                              hipStream_t stream) {
    const float* x    = (const float*)d_in[0];
    const float* x90  = (const float*)d_in[1];
    const float* x180 = (const float*)d_in[2];
    const float* x270 = (const float*)d_in[3];
    const float* W_f  = (const float*)d_in[4];
    const float* b_f  = (const float*)d_in[5];
    const float* W_r  = (const float*)d_in[6];
    const float* b_r  = (const float*)d_in[7];
    const float* cent = (const float*)d_in[8];
    float* out = (float*)d_out;

    // Workspace layout (~57 MB total)
    bf16_t* xbf  = (bf16_t*)d_ws;                         // 4 * 4096*1024 bf16
    bf16_t* wbf  = xbf + (size_t)4 * B_SZ * IN_FT;        // 2 * 512*1024  bf16
    bf16_t* feat = wbf + (size_t)2 * FEAT * IN_FT;        // 4096*2048     bf16
    bf16_t* cbf  = feat + (size_t)B_SZ * D_F;             // 1008*2048     bf16
    float*  f2   = (float*)(cbf + (size_t)NPAD * D_F);    // 4096 f32
    float*  c2   = f2 + B_SZ;                             // 1008 f32

    const int nx = B_SZ * IN_FT;          // per x-array elements
    const int nw = FEAT * IN_FT;          // per weight-matrix elements
    const int gx = nx / 8 / 256;          // 2048 blocks (exact)
    const int gw = nw / 8 / 256;          // 256 blocks  (exact)

    cvt_bf16<<<gx, 256, 0, stream>>>(x,    xbf + (size_t)0 * nx, nx);
    cvt_bf16<<<gx, 256, 0, stream>>>(x90,  xbf + (size_t)1 * nx, nx);
    cvt_bf16<<<gx, 256, 0, stream>>>(x180, xbf + (size_t)2 * nx, nx);
    cvt_bf16<<<gx, 256, 0, stream>>>(x270, xbf + (size_t)3 * nx, nx);
    cvt_bf16<<<gw, 256, 0, stream>>>(W_f,  wbf + (size_t)0 * nw, nw);
    cvt_bf16<<<gw, 256, 0, stream>>>(W_r,  wbf + (size_t)1 * nw, nw);
    prep_centers<<<NPAD, 256, 0, stream>>>(cent, cbf, c2);
    zero_f2<<<(B_SZ + 255) / 256, 256, 0, stream>>>(f2);

    feature_gemm<<<dim3(B_SZ / 128, FEAT / 64, 4), 128, 0, stream>>>(
        xbf, wbf, b_f, b_r, feat, f2);
    dist_gemm<<<dim3(B_SZ / 128, NPAD / 16), 128, 0, stream>>>(feat, cbf, f2, c2, out);
}
